// ResMHAtten_85444079387120
// MI455X (gfx1250) — matrix operand order repaired and checked by reading, not compiled, not run on hardware
//
#include <hip/hip_runtime.h>

// ---------- types ----------
typedef __attribute__((ext_vector_type(16))) __bf16         v16bf;
typedef __attribute__((ext_vector_type(16))) unsigned short v16us;
typedef __attribute__((ext_vector_type(8)))  float          v8f;
typedef __attribute__((ext_vector_type(4)))  unsigned int   v4ui;
typedef __attribute__((ext_vector_type(4)))  int            v4si;
typedef __attribute__((ext_vector_type(8)))  int            v8si;

#define BB 4
#define TT 2048
#define EE 1024
#define HH 16
#define HS 64

#if defined(__has_builtin)
#if __has_builtin(__builtin_amdgcn_tensor_load_to_lds)
#define HAVE_TDM 1
#endif
#endif
#ifndef HAVE_TDM
#define HAVE_TDM 0
#endif

__device__ __forceinline__ unsigned short f2bf_bits(float f) {
  unsigned int u = __float_as_uint(f);
  u += 0x7FFFu + ((u >> 16) & 1u);          // round-to-nearest-even
  return (unsigned short)(u >> 16);
}

__device__ __forceinline__ v8f vzero() {
  v8f z;
#pragma unroll
  for (int i = 0; i < 8; ++i) z[i] = 0.0f;
  return z;
}

__device__ __forceinline__ v8f wmma_bf16(v16bf a, v16bf b, v8f c) {
  return __builtin_amdgcn_wmma_f32_16x16x32_bf16(
      false, a, false, b, (short)0, c, false, false);
}

// A-fragment: 16x32 bf16, row m = lane&15, K halves split across lane halves.
__device__ __forceinline__ v16bf frag_a(const unsigned short* p, int ld) {
  const int lane = threadIdx.x & 31;
  const int m = lane & 15, hf = lane >> 4;
  v16us t;
#pragma unroll
  for (int v = 0; v < 8; ++v) {
    const int kb = (hf << 4) + (v << 1);
    t[2 * v]     = p[m * ld + kb];
    t[2 * v + 1] = p[m * ld + kb + 1];
  }
  return __builtin_bit_cast(v16bf, t);
}

// B-fragment: 32x16 bf16, B(k,n) = p[k*ld + n]; n = lane&15, K halves split.
__device__ __forceinline__ v16bf frag_b(const unsigned short* p, int ld) {
  const int lane = threadIdx.x & 31;
  const int n = lane & 15, hf = lane >> 4;
  v16us t;
#pragma unroll
  for (int v = 0; v < 8; ++v) {
    const int k = (hf << 4) + (v << 1);
    t[2 * v]     = p[k * ld + n];
    t[2 * v + 1] = p[(k + 1) * ld + n];
  }
  return __builtin_bit_cast(v16bf, t);
}

// Transposed B-fragment: B(k,n) = p[n*ld + k]  (for Q @ K^T).
__device__ __forceinline__ v16bf frag_bT(const unsigned short* p, int ld) {
  const int lane = threadIdx.x & 31;
  const int n = lane & 15, hf = lane >> 4;
  v16us t;
#pragma unroll
  for (int v = 0; v < 8; ++v) {
    const int k = (hf << 4) + (v << 1);
    t[2 * v]     = p[n * ld + k];
    t[2 * v + 1] = p[n * ld + k + 1];
  }
  return __builtin_bit_cast(v16bf, t);
}

#if HAVE_TDM
// TDM: DMA a 2D bf16 tile (tile_d0 x tile_d1) from global into LDS, with
// optional LDS row padding (pad codes per D# group1 spec).
// D# group0: count=1 | lds_addr | global_addr | type=2.
// D# group1: data_size=2B, pad fields, tensor dims, tile dims, dim0 stride.
__device__ __forceinline__ void tdm_load_2d(unsigned lds_off,
                                            const unsigned short* g,
                                            unsigned tile_d0, unsigned tile_d1,
                                            unsigned tensor_d0,
                                            unsigned tensor_d1,
                                            unsigned stride_d0,
                                            unsigned pad_interval_code,
                                            unsigned pad_amount_code) {
  const unsigned long long ga = (unsigned long long)(size_t)g;
  v4ui g0;
  g0[0] = 1u;                                   // count=1 (valid user D#)
  g0[1] = lds_off;                              // lds_addr (bytes)
  g0[2] = (unsigned)ga;                         // global_addr[31:0]
  g0[3] = (unsigned)(ga >> 32) | (2u << 30);    // addr[56:32] | type=2
  const unsigned pad_en = (pad_interval_code | pad_amount_code) ? 1u : 0u;
  v8si g1;
  g1[0] = (int)((1u << 16) | (pad_en << 20) | (pad_interval_code << 22) |
                (pad_amount_code << 25));       // data_size=2B + pad cfg
  g1[1] = (int)((tensor_d0 & 0xFFFFu) << 16);   // tensor_dim0[15:0]
  g1[2] = (int)((tensor_d0 >> 16) | ((tensor_d1 & 0xFFFFu) << 16));
  g1[3] = (int)((tensor_d1 >> 16) | (tile_d0 << 16));  // tile_dim0
  g1[4] = (int)(tile_d1 & 0xFFFFu);             // tile_dim1 (tile_dim2=0)
  g1[5] = (int)stride_d0;                       // tensor_dim0_stride[31:0]
  g1[6] = 0;
  g1[7] = 0;
  v4si z4;
#pragma unroll
  for (int i = 0; i < 4; ++i) z4[i] = 0;
#if __clang_major__ >= 23
  v8si z8;
#pragma unroll
  for (int i = 0; i < 8; ++i) z8[i] = 0;
  __builtin_amdgcn_tensor_load_to_lds(g0, g1, z4, z4, z8, 0);
#else
  __builtin_amdgcn_tensor_load_to_lds(g0, g1, z4, z4, 0);
#endif
}
#endif  // HAVE_TDM

// ---------- kernel 1: fp32 -> bf16 ----------
__global__ __launch_bounds__(256) void cvt_bf16_k(const float* __restrict__ s,
                                                  unsigned short* __restrict__ d,
                                                  int n) {
  const int i = blockIdx.x * 256 + threadIdx.x;
  if (i < n) d[i] = f2bf_bits(s[i]);
}

// ---------- kernel 2: LayerNorm (E=1024) -> bf16 ----------
__global__ __launch_bounds__(256) void ln_bf16_k(const float* __restrict__ x,
                                                 const float* __restrict__ g,
                                                 const float* __restrict__ bta,
                                                 unsigned short* __restrict__ xn) {
  const int row = blockIdx.x;
  const int tid = threadIdx.x;
  const float* xr = x + (size_t)row * EE;
  __shared__ float red[16];
  float v0[4];
  float s = 0.f;
#pragma unroll
  for (int i = 0; i < 4; ++i) { v0[i] = xr[tid + i * 256]; s += v0[i]; }
#pragma unroll
  for (int m = 16; m >= 1; m >>= 1) s += __shfl_xor(s, m, 32);
  if ((tid & 31) == 0) red[tid >> 5] = s;
  __syncthreads();
  float tot = 0.f;
#pragma unroll
  for (int i = 0; i < 8; ++i) tot += red[i];
  const float mu = tot * (1.0f / 1024.0f);
  float vs = 0.f;
#pragma unroll
  for (int i = 0; i < 4; ++i) { const float d = v0[i] - mu; vs += d * d; }
#pragma unroll
  for (int m = 16; m >= 1; m >>= 1) vs += __shfl_xor(vs, m, 32);
  if ((tid & 31) == 0) red[8 + (tid >> 5)] = vs;
  __syncthreads();
  float tv = 0.f;
#pragma unroll
  for (int i = 0; i < 8; ++i) tv += red[8 + i];
  const float rstd = rsqrtf(tv * (1.0f / 1024.0f) + 1e-5f);
#pragma unroll
  for (int i = 0; i < 4; ++i) {
    const int c = tid + i * 256;
    xn[(size_t)row * EE + c] = f2bf_bits((v0[i] - mu) * rstd * g[c] + bta[c]);
  }
}

// ---------- kernel 3: tiled WMMA GEMM, 128x128 block tile ----------
// MODE 0: Q proj  -> scatter bf16 [B,H,T,HS], *1/sqrt(HS) folded in
// MODE 1: KV proj -> scatter bf16 K/V [B,H,T,HS]
// MODE 2: out proj -> fp32 out + bias + residual
template <int MODE>
__global__ __launch_bounds__(256) void gemm_wmma(
    const unsigned short* __restrict__ A,   // [8192, 1024] bf16
    const unsigned short* __restrict__ Bw,  // [1024, N] bf16
    const float* __restrict__ bias,
    const float* __restrict__ resid,
    float* __restrict__ outF,
    unsigned short* __restrict__ outK,
    unsigned short* __restrict__ outV,
    int N) {
  constexpr int K = EE;
  const int m0 = blockIdx.y * 128;
  const int n0 = blockIdx.x * 128;
  const int tid = threadIdx.x;
  const int wid = tid >> 5;
  const int lane = tid & 31;
  const int wm = (wid >> 1) << 5;   // 0,32,64,96
  const int wn = (wid & 1) << 6;    // 0,64
  v8f acc[2][4];
#pragma unroll
  for (int i = 0; i < 2; ++i)
#pragma unroll
    for (int j = 0; j < 4; ++j) acc[i][j] = vzero();

#if HAVE_TDM
  // Double-buffered TDM pipeline: wave 0 DMAs step k+1 while all waves
  // run WMMA on step k.  A tile: 128x32 (LDS ld 36 via pad 16dw->+2dw).
  // B tile: 32x128 (LDS ld 132 via pad 64dw->+2dw).
  __shared__ unsigned short As[2][128 * 36];
  __shared__ unsigned short Bs[2][32 * 132];
  const unsigned short* Abase = A + (size_t)m0 * K;
  const unsigned short* Bbase = Bw + n0;
  if (wid == 0) {
    tdm_load_2d((unsigned)(size_t)&As[0][0], Abase, 32u, 128u,
                (unsigned)K, 8192u, (unsigned)K, 3u, 1u);
    tdm_load_2d((unsigned)(size_t)&Bs[0][0], Bbase, 128u, 32u,
                (unsigned)N, (unsigned)K, (unsigned)N, 5u, 1u);
  }
  for (int k0 = 0; k0 < K; k0 += 32) {
    const int cur = (k0 >> 5) & 1;
    if (wid == 0) __builtin_amdgcn_s_wait_tensorcnt(0);
    __syncthreads();
    if (wid == 0 && (k0 + 32 < K)) {
      tdm_load_2d((unsigned)(size_t)&As[cur ^ 1][0], Abase + (k0 + 32),
                  32u, 128u, (unsigned)K, 8192u, (unsigned)K, 3u, 1u);
      tdm_load_2d((unsigned)(size_t)&Bs[cur ^ 1][0],
                  Bbase + (size_t)(k0 + 32) * N, 128u, 32u,
                  (unsigned)N, (unsigned)K, (unsigned)N, 5u, 1u);
    }
    v16bf af[2], bfm[4];
#pragma unroll
    for (int i = 0; i < 2; ++i)
      af[i] = frag_a(&As[cur][(wm + (i << 4)) * 36], 36);
#pragma unroll
    for (int j = 0; j < 4; ++j) bfm[j] = frag_b(&Bs[cur][wn + (j << 4)], 132);
#pragma unroll
    for (int i = 0; i < 2; ++i)
#pragma unroll
      for (int j = 0; j < 4; ++j) acc[i][j] = wmma_bf16(af[i], bfm[j], acc[i][j]);
  }
#else
  __shared__ unsigned short As[128 * 36];
  __shared__ unsigned short Bs[32 * 132];
  for (int k0 = 0; k0 < K; k0 += 32) {
    __syncthreads();
    const unsigned int* Ag = (const unsigned int*)(A + (size_t)m0 * K + k0);
    const unsigned int* Bg = (const unsigned int*)(Bw + (size_t)k0 * N + n0);
#pragma unroll
    for (int it = 0; it < 8; ++it) {
      const int i = tid + (it << 8);
      {
        const int r = i >> 4, c = (i & 15) << 1;
        *(unsigned int*)&As[r * 36 + c] = Ag[(size_t)r * (K >> 1) + (i & 15)];
      }
      {
        const int r = i >> 6, c = (i & 63) << 1;
        *(unsigned int*)&Bs[r * 132 + c] = Bg[(size_t)r * (N >> 1) + (i & 63)];
      }
    }
    if (k0 + 32 < K)
      __builtin_prefetch(A + (size_t)(m0 + (tid >> 1)) * K + k0 + 32, 0, 0);
    __syncthreads();
    v16bf af[2], bfm[4];
#pragma unroll
    for (int i = 0; i < 2; ++i) af[i] = frag_a(&As[(wm + (i << 4)) * 36], 36);
#pragma unroll
    for (int j = 0; j < 4; ++j) bfm[j] = frag_b(&Bs[wn + (j << 4)], 132);
#pragma unroll
    for (int i = 0; i < 2; ++i)
#pragma unroll
      for (int j = 0; j < 4; ++j) acc[i][j] = wmma_bf16(af[i], bfm[j], acc[i][j]);
  }
#endif

  const int hf = lane >> 4, nn = lane & 15;
#pragma unroll
  for (int i = 0; i < 2; ++i) {
#pragma unroll
    for (int j = 0; j < 4; ++j) {
#pragma unroll
      for (int v = 0; v < 8; ++v) {
        const int m = m0 + wm + (i << 4) + v + (hf << 3);
        const int n = n0 + wn + (j << 4) + nn;
        float x = acc[i][j][v];
        if (MODE == 0) {
          x = (x + bias[n]) * 0.125f;  // fold 1/sqrt(HS)
          const int b = m >> 11, t = m & 2047, h = n >> 6, hs = n & 63;
          outK[(((size_t)(b * HH + h) * TT) + t) * HS + hs] = f2bf_bits(x);
        } else if (MODE == 1) {
          x += bias[n];
          const int b = m >> 11, t = m & 2047;
          if (n < 1024) {
            const int h = n >> 6, hs = n & 63;
            outK[(((size_t)(b * HH + h) * TT) + t) * HS + hs] = f2bf_bits(x);
          } else {
            const int h = (n - 1024) >> 6, hs = n & 63;
            outV[(((size_t)(b * HH + h) * TT) + t) * HS + hs] = f2bf_bits(x);
          }
        } else {
          outF[(size_t)m * EE + n] = x + bias[n] + resid[(size_t)m * EE + n];
        }
      }
    }
  }
}

// ---------- kernel 4: flash attention, causal, per (b,h) 128-query tile ----
__global__ __launch_bounds__(256) void attn_wmma(
    const unsigned short* __restrict__ Qb,  // [B,H,T,HS] bf16 (pre-scaled)
    const unsigned short* __restrict__ Kb,
    const unsigned short* __restrict__ Vb,
    unsigned short* __restrict__ ctx) {     // [B*T, H*HS] bf16
  const int qt = blockIdx.x;                // 0..15
  const int bh = blockIdx.y;                // 0..63
  const int b = bh >> 4, h = bh & 15;
  const int q_base = qt << 7;
  __shared__ unsigned short Qs[128 * 72];
  __shared__ unsigned short Ks[64 * 72];
  __shared__ unsigned short Vs[64 * 72];
  __shared__ unsigned short Ps[128 * 72];
  const int tid = threadIdx.x, wid = tid >> 5, lane = tid & 31;
  const int hf = lane >> 4, nn = lane & 15;
  const size_t base = (size_t)bh * TT * HS;

#if HAVE_TDM
  // Q tile 128x64 contiguous rows; LDS ld 72 via pad 32dw -> +4dw.
  if (wid == 0)
    tdm_load_2d((unsigned)(size_t)Qs, Qb + base + (size_t)q_base * HS,
                64u, 128u, 64u, (unsigned)TT, 64u, 4u, 3u);
#else
#pragma unroll
  for (int it = 0; it < 16; ++it) {
    const int i = tid + (it << 8);
    const int r = i >> 5, c = (i & 31) << 1;
    *(unsigned int*)&Qs[r * 72 + c] =
        *(const unsigned int*)&Qb[base + (size_t)(q_base + r) * HS + c];
  }
#endif

  v8f O[4];
#pragma unroll
  for (int j = 0; j < 4; ++j) O[j] = vzero();
  float mr[8], lr[8];
#pragma unroll
  for (int v = 0; v < 8; ++v) { mr[v] = -1e30f; lr[v] = 0.f; }

  const int nkt = (qt << 1) + 2;
  for (int kt = 0; kt < nkt; ++kt) {
    __syncthreads();
#if HAVE_TDM
    if (wid == 0) {
      tdm_load_2d((unsigned)(size_t)Ks, Kb + base + (size_t)(kt << 6) * HS,
                  64u, 64u, 64u, (unsigned)TT, 64u, 4u, 3u);
      tdm_load_2d((unsigned)(size_t)Vs, Vb + base + (size_t)(kt << 6) * HS,
                  64u, 64u, 64u, (unsigned)TT, 64u, 4u, 3u);
      __builtin_amdgcn_s_wait_tensorcnt(0);
    }
#else
#pragma unroll
    for (int it = 0; it < 8; ++it) {
      const int i = tid + (it << 8);
      const int r = i >> 5, c = (i & 31) << 1;
      const size_t g = base + (size_t)((kt << 6) + r) * HS + c;
      *(unsigned int*)&Ks[r * 72 + c] = *(const unsigned int*)&Kb[g];
      *(unsigned int*)&Vs[r * 72 + c] = *(const unsigned int*)&Vb[g];
    }
#endif
    __syncthreads();

    // S = Q K^T for this wave's 16 rows x 64 keys
    v8f s[4];
#pragma unroll
    for (int j = 0; j < 4; ++j) {
      s[j] = vzero();
#pragma unroll
      for (int kk = 0; kk < HS; kk += 32) {
        const v16bf a = frag_a(&Qs[(wid << 4) * 72 + kk], 72);
        const v16bf bt = frag_bT(&Ks[(j << 4) * 72 + kk], 72);
        s[j] = wmma_bf16(a, bt, s[j]);
      }
    }

    // causal mask + online softmax (row = v + 8*hf, replicated over 16 lanes)
#pragma unroll
    for (int v = 0; v < 8; ++v) {
      const int m = v + (hf << 3);
      const int qr = q_base + (wid << 4) + m;
      float rmax = -1e30f;
#pragma unroll
      for (int j = 0; j < 4; ++j) {
        const int kc = (kt << 6) + (j << 4) + nn;
        if (kc > qr) s[j][v] = -1e30f;
        rmax = fmaxf(rmax, s[j][v]);
      }
      rmax = fmaxf(rmax, __shfl_xor(rmax, 1, 32));
      rmax = fmaxf(rmax, __shfl_xor(rmax, 2, 32));
      rmax = fmaxf(rmax, __shfl_xor(rmax, 4, 32));
      rmax = fmaxf(rmax, __shfl_xor(rmax, 8, 32));
      const float mnew = fmaxf(mr[v], rmax);
      const float corr = __expf(mr[v] - mnew);
      float rsum = 0.f;
#pragma unroll
      for (int j = 0; j < 4; ++j) {
        const float p = __expf(s[j][v] - mnew);
        rsum += p;
        Ps[((wid << 4) + m) * 72 + (j << 4) + nn] = f2bf_bits(p);
      }
      rsum += __shfl_xor(rsum, 1, 32);
      rsum += __shfl_xor(rsum, 2, 32);
      rsum += __shfl_xor(rsum, 4, 32);
      rsum += __shfl_xor(rsum, 8, 32);
      lr[v] = lr[v] * corr + rsum;
      mr[v] = mnew;
#pragma unroll
      for (int j = 0; j < 4; ++j) O[j][v] *= corr;
    }

    // O += P @ V
#pragma unroll
    for (int j = 0; j < 4; ++j) {
#pragma unroll
      for (int kk = 0; kk < 64; kk += 32) {
        const v16bf a = frag_a(&Ps[(wid << 4) * 72 + kk], 72);
        const v16bf bv = frag_b(&Vs[kk * 72 + (j << 4)], 72);
        O[j] = wmma_bf16(a, bv, O[j]);
      }
    }
  }

  // normalize and write ctx as [B*T, H*HS] bf16
#pragma unroll
  for (int v = 0; v < 8; ++v) {
    const int m = v + (hf << 3);
    const int t = q_base + (wid << 4) + m;
    const float inv = 1.0f / lr[v];
#pragma unroll
    for (int j = 0; j < 4; ++j)
      ctx[((size_t)(b * TT + t)) * EE + (h << 6) + (j << 4) + nn] =
          f2bf_bits(O[j][v] * inv);
  }
}

// ---------- launch ----------
extern "C" void kernel_launch(void* const* d_in, const int* in_sizes, int n_in,
                              void* d_out, int out_size, void* d_ws,
                              size_t ws_size, hipStream_t stream) {
  (void)in_sizes; (void)n_in; (void)out_size; (void)ws_size;
  const float* q    = (const float*)d_in[0];
  // d_in[1] == kv == q (self-attention), unused
  const float* ln_g = (const float*)d_in[2];
  const float* ln_b = (const float*)d_in[3];
  const float* wq   = (const float*)d_in[4];
  const float* bq   = (const float*)d_in[5];
  const float* wkv  = (const float*)d_in[6];
  const float* bkv  = (const float*)d_in[7];
  const float* wo   = (const float*)d_in[8];
  const float* bo   = (const float*)d_in[9];
  float* out = (float*)d_out;

  const size_t MTOK = (size_t)BB * TT;  // 8192
  char* w = (char*)d_ws;
  unsigned short* xn   = (unsigned short*)w; w += MTOK * EE * 2;
  unsigned short* wqb  = (unsigned short*)w; w += (size_t)EE * EE * 2;
  unsigned short* wkvb = (unsigned short*)w; w += (size_t)EE * 2 * EE * 2;
  unsigned short* wob  = (unsigned short*)w; w += (size_t)EE * EE * 2;
  unsigned short* Qb   = (unsigned short*)w; w += MTOK * EE * 2;
  unsigned short* Kb   = (unsigned short*)w; w += MTOK * EE * 2;
  unsigned short* Vb   = (unsigned short*)w; w += MTOK * EE * 2;
  unsigned short* ctx  = (unsigned short*)w; w += MTOK * EE * 2;

  cvt_bf16_k<<<(EE * EE + 255) / 256, 256, 0, stream>>>(wq, wqb, EE * EE);
  cvt_bf16_k<<<(EE * 2 * EE + 255) / 256, 256, 0, stream>>>(wkv, wkvb,
                                                            EE * 2 * EE);
  cvt_bf16_k<<<(EE * EE + 255) / 256, 256, 0, stream>>>(wo, wob, EE * EE);

  ln_bf16_k<<<(int)MTOK, 256, 0, stream>>>(q, ln_g, ln_b, xn);

  gemm_wmma<0><<<dim3(8, 64), 256, 0, stream>>>(xn, wqb, bq, nullptr, nullptr,
                                                Qb, nullptr, EE);
  gemm_wmma<1><<<dim3(16, 64), 256, 0, stream>>>(xn, wkvb, bkv, nullptr,
                                                 nullptr, Kb, Vb, 2 * EE);

  attn_wmma<<<dim3(TT / 128, BB * HH), 256, 0, stream>>>(Qb, Kb, Vb, ctx);

  gemm_wmma<2><<<dim3(8, 64), 256, 0, stream>>>(ctx, wob, bo, q, out, nullptr,
                                                nullptr, EE);
}
